// GroundingModule_60009283059965
// MI455X (gfx1250) — matrix operand order repaired and checked by reading, not compiled, not run on hardware
//
#include <hip/hip_runtime.h>

// ---------------------------------------------------------------------------
// Grounding module for MI455X (gfx1250, wave32, WMMA f32_16x16x32_f16)
// BS=4096, L=32, D=768, H=384
// ---------------------------------------------------------------------------

typedef __attribute__((ext_vector_type(16))) _Float16 v16h;
typedef __attribute__((ext_vector_type(8)))  float    v8f;

#define D_DIM 768
#define H_DIM 384
#define L_DIM 32
#define BS    4096
#define KB    24      // 768 / 32 k-blocks
#define NT1   48      // 768 / 16 n-tiles (GEMM1)
#define NT2   24      // 384 / 16 n-tiles (GEMM2)
#define ROWS  64      // rows per block in main kernel (2 batches)

// output layout (concatenated flat, reference return order)
#define OFF_GS   0
#define OFF_MIDX (BS * L_DIM)                 // 131072
#define OFF_ST   (OFF_MIDX + BS * 2)          // 139264
#define OFF_EN   (OFF_ST + BS)                // 143360
#define OFF_MASK (OFF_EN + BS)                // 147456
#define OFF_OK   (OFF_MASK + BS * L_DIM)      // 278528

__device__ __forceinline__ float geluf(float x) {
  return 0.5f * x * (1.0f + erff(x * 0.70710678118654752440f));
}

__device__ __forceinline__ float wsum(float v) {
#pragma unroll
  for (int o = 16; o >= 1; o >>= 1) v += __shfl_xor(v, o, 32);
  return v;
}
__device__ __forceinline__ float wmaxr(float v) {
#pragma unroll
  for (int o = 16; o >= 1; o >>= 1) v = fmaxf(v, __shfl_xor(v, o, 32));
  return v;
}

__device__ __forceinline__ v8f wmma_f16(v16h a, v16h b, v8f c) {
  return __builtin_amdgcn_wmma_f32_16x16x32_f16(false, a, false, b, (short)0, c,
                                                false, false);
}

// ---------------------------------------------------------------------------
// Swizzle + f32->f16 convert weights into the per-lane WMMA B operand layout.
// B fragment (32x16 KxN f16): lane = (N%16) | (K>=16)<<4 ; slot j2 = K%16.
// Tile t = kb*NT + nt; each lane owns 16 contiguous f16 (=> b128x2 loads).
// ---------------------------------------------------------------------------
__global__ __launch_bounds__(256) void swizzleWKernel(
    const float* __restrict__ W, _Float16* __restrict__ out, int K, int N) {
  const int wv = threadIdx.x >> 5, lane = threadIdx.x & 31;
  const int NT = N >> 4, KBt = K >> 5;
  const int t = blockIdx.x * 8 + wv;
  if (t >= NT * KBt) return;
  const int kb = t / NT, nt = t % NT;
  const int kbase = kb * 32 + ((lane >> 4) << 4);
  const int n = nt * 16 + (lane & 15);
  _Float16* o = out + ((size_t)t * 32 + lane) * 16;
  const float* w = W + (size_t)kbase * N + n;
#pragma unroll
  for (int j = 0; j < 16; ++j) o[j] = (_Float16)w[(size_t)j * N];
}

// A fragment swizzle constants (16x32 MxK f16, contiguous k order used by this code):
//   lane = (M%16) | ((kk>>4)&1)<<4 ; slot = kk&15
// With per-lane k = lane + 32*i  =>  kk == lane, so both are lane-constant.

// ---------------------------------------------------------------------------
// qa path: qa2 = gelu(LN(qa) @ qp_W + qp_bias), stored f16 (natural layout).
// One block = 64 rows (M=64), 8 waves; dyn LDS = 64*768 f16 = 96 KB.
// ---------------------------------------------------------------------------
__global__ __launch_bounds__(256) void qaProjKernel(
    const float* __restrict__ qa, const float* __restrict__ qg,
    const float* __restrict__ qb, const _Float16* __restrict__ bsw,
    const float* __restrict__ qbias, _Float16* __restrict__ qa2) {
  extern __shared__ char smem[];
  _Float16* aSw = (_Float16*)smem;
  const int wv = threadIdx.x >> 5, lane = threadIdx.x & 31;
  const int m0 = blockIdx.x * ROWS;
  const int lane_off = ((lane >> 4) & 1) << 4;
  const int slot = lane & 15;

  float gk[KB], bk[KB];
#pragma unroll
  for (int i = 0; i < KB; ++i) {
    gk[i] = qg[lane + 32 * i];
    bk[i] = qb[lane + 32 * i];
  }
  // LN(qa rows) -> swizzled f16 A in LDS
  for (int rr = 0; rr < 8; ++rr) {
    const int row = wv * 8 + rr;
    const float* xr = qa + (size_t)(m0 + row) * D_DIM;
    float xv[KB], s = 0.f, s2 = 0.f;
#pragma unroll
    for (int i = 0; i < KB; ++i) {
      float x = xr[lane + 32 * i];
      xv[i] = x; s += x; s2 += x * x;
    }
    s = wsum(s); s2 = wsum(s2);
    const float mean = s * (1.f / D_DIM);
    const float rs = rsqrtf(s2 * (1.f / D_DIM) - mean * mean + 1e-5f);
    const int mt = row >> 4, lsw = (row & 15) | lane_off;
#pragma unroll
    for (int i = 0; i < KB; ++i)
      aSw[(((mt * KB) + i) * 32 + lsw) * 16 + slot] =
          (_Float16)((xv[i] - mean) * rs * gk[i] + bk[i]);
  }
  __syncthreads();

  v8f acc[4][6] = {};
  for (int kb = 0; kb < KB; ++kb) {
    v16h a[4];
#pragma unroll
    for (int mt = 0; mt < 4; ++mt)
      a[mt] = *(const v16h*)&aSw[(((mt * KB) + kb) * 32 + lane) * 16];
#pragma unroll
    for (int t = 0; t < 6; ++t) {
      const int nt = wv * 6 + t;
      v16h bf = *(const v16h*)&bsw[(((size_t)kb * NT1 + nt) * 32 + lane) * 16];
#pragma unroll
      for (int mt = 0; mt < 4; ++mt) acc[mt][t] = wmma_f16(a[mt], bf, acc[mt][t]);
    }
  }
  // epilogue: bias + exact GELU -> f16
#pragma unroll
  for (int t = 0; t < 6; ++t) {
    const int n = (wv * 6 + t) * 16 + (lane & 15);
    const float bb = qbias[n];
#pragma unroll
    for (int mt = 0; mt < 4; ++mt)
#pragma unroll
      for (int r = 0; r < 8; ++r) {
        const int row = mt * 16 + r + ((lane >> 4) << 3);
        qa2[(size_t)(m0 + row) * D_DIM + n] = (_Float16)geluf(acc[mt][t][r] + bb);
      }
  }
}

// ---------------------------------------------------------------------------
// Fused main kernel: one block = 2 batches (64 rows), 8 waves.
// dyn LDS: aSw (64x768 f16, reused as h_sw) + v2s (64x768 f16, reused as
// h2s 64x384 f32) + logits (64 f32)  ~= 192.25 KB (CDNA5 WGP has 320 KB).
// ---------------------------------------------------------------------------
__global__ __launch_bounds__(256) void groundingMainKernel(
    const float* __restrict__ v, const float* __restrict__ vp_g,
    const float* __restrict__ vp_b, const _Float16* __restrict__ bswVP,
    const float* __restrict__ vp_bias, const float* __restrict__ gr_g,
    const float* __restrict__ gr_b, const _Float16* __restrict__ bswG1,
    const float* __restrict__ gr_b1, const float* __restrict__ gr_W2,
    const float* __restrict__ gr_b2, const _Float16* __restrict__ qa2,
    const float* __restrict__ sigmaP, float* __restrict__ out) {
  extern __shared__ char smem[];
  _Float16* aSw = (_Float16*)smem;                               // 64*768 f16
  _Float16* v2s = (_Float16*)(smem + ROWS * D_DIM * 2);          // 64*768 f16
  float* h2s = (float*)(smem + ROWS * D_DIM * 2);                // alias: 64*384 f32
  float* logitsS = (float*)(smem + 2 * ROWS * D_DIM * 2);        // 64 f32

  const int wv = threadIdx.x >> 5, lane = threadIdx.x & 31;
  const int blk = blockIdx.x;
  const int rowBase = wv * 8;
  const int lane_off = ((lane >> 4) & 1) << 4;
  const int slot = lane & 15;

  // ---------------- pass 1: LN(v) -> swizzled f16 A ----------------
  {
    float gk[KB], bk[KB];
#pragma unroll
    for (int i = 0; i < KB; ++i) {
      gk[i] = vp_g[lane + 32 * i];
      bk[i] = vp_b[lane + 32 * i];
    }
    for (int rr = 0; rr < 8; ++rr) {
      const int row = rowBase + rr;
      const float* vr = v + ((size_t)blk * ROWS + row) * D_DIM;
      float xv[KB], s = 0.f, s2 = 0.f;
#pragma unroll
      for (int i = 0; i < KB; ++i) {
        float x = vr[lane + 32 * i];
        xv[i] = x; s += x; s2 += x * x;
      }
      s = wsum(s); s2 = wsum(s2);
      const float mean = s * (1.f / D_DIM);
      const float rs = rsqrtf(s2 * (1.f / D_DIM) - mean * mean + 1e-5f);
      const int mt = row >> 4, lsw = (row & 15) | lane_off;
#pragma unroll
      for (int i = 0; i < KB; ++i)
        aSw[(((mt * KB) + i) * 32 + lsw) * 16 + slot] =
            (_Float16)((xv[i] - mean) * rs * gk[i] + bk[i]);
    }
  }
  __syncthreads();

  // ---------------- GEMM1: v2 = gelu(A @ vp_W + bias) ----------------
  {
    v8f acc[4][6] = {};
    for (int kb = 0; kb < KB; ++kb) {
      v16h a[4];
#pragma unroll
      for (int mt = 0; mt < 4; ++mt)
        a[mt] = *(const v16h*)&aSw[(((mt * KB) + kb) * 32 + lane) * 16];
#pragma unroll
      for (int t = 0; t < 6; ++t) {
        const int nt = wv * 6 + t;
        v16h bf = *(const v16h*)&bswVP[(((size_t)kb * NT1 + nt) * 32 + lane) * 16];
#pragma unroll
        for (int mt = 0; mt < 4; ++mt) acc[mt][t] = wmma_f16(a[mt], bf, acc[mt][t]);
      }
    }
#pragma unroll
    for (int t = 0; t < 6; ++t) {
      const int n = (wv * 6 + t) * 16 + (lane & 15);
      const float bb = vp_bias[n];
#pragma unroll
      for (int mt = 0; mt < 4; ++mt)
#pragma unroll
        for (int r = 0; r < 8; ++r) {
          const int row = mt * 16 + r + ((lane >> 4) << 3);
          v2s[(size_t)row * D_DIM + n] = (_Float16)geluf(acc[mt][t][r] + bb);
        }
    }
  }
  __syncthreads();

  // ------- gate = tanh(v2 . qa2); h = LN(v2*gate) -> swizzled A -------
  {
    float gk[KB], bk[KB];
#pragma unroll
    for (int i = 0; i < KB; ++i) {
      gk[i] = gr_g[lane + 32 * i];
      bk[i] = gr_b[lane + 32 * i];
    }
    for (int rr = 0; rr < 8; ++rr) {
      const int row = rowBase + rr;
      const size_t gb = (size_t)blk * 2 + (row >> 5);
      const _Float16* qr = qa2 + gb * D_DIM;
      const _Float16* v2r = v2s + (size_t)row * D_DIM;
      float dot = 0.f;
#pragma unroll
      for (int i = 0; i < KB; ++i)
        dot += (float)v2r[lane + 32 * i] * (float)qr[lane + 32 * i];
      const float gate = tanhf(wsum(dot));
      float yv[KB], s = 0.f, s2 = 0.f;
#pragma unroll
      for (int i = 0; i < KB; ++i) {
        float y = (float)v2r[lane + 32 * i] * gate;
        yv[i] = y; s += y; s2 += y * y;
      }
      s = wsum(s); s2 = wsum(s2);
      const float mean = s * (1.f / D_DIM);
      const float rs = rsqrtf(s2 * (1.f / D_DIM) - mean * mean + 1e-5f);
      const int mt = row >> 4, lsw = (row & 15) | lane_off;
#pragma unroll
      for (int i = 0; i < KB; ++i)
        aSw[(((mt * KB) + i) * 32 + lsw) * 16 + slot] =
            (_Float16)((yv[i] - mean) * rs * gk[i] + bk[i]);
    }
  }
  __syncthreads();

  // ---------------- GEMM2: h2 = gelu(h @ gr_W1 + b1) ----------------
  {
    v8f acc[4][3] = {};
    for (int kb = 0; kb < KB; ++kb) {
      v16h a[4];
#pragma unroll
      for (int mt = 0; mt < 4; ++mt)
        a[mt] = *(const v16h*)&aSw[(((mt * KB) + kb) * 32 + lane) * 16];
#pragma unroll
      for (int t = 0; t < 3; ++t) {
        const int nt = wv * 3 + t;
        v16h bf = *(const v16h*)&bswG1[(((size_t)kb * NT2 + nt) * 32 + lane) * 16];
#pragma unroll
        for (int mt = 0; mt < 4; ++mt) acc[mt][t] = wmma_f16(a[mt], bf, acc[mt][t]);
      }
    }
#pragma unroll
    for (int t = 0; t < 3; ++t) {
      const int n = (wv * 3 + t) * 16 + (lane & 15);
      const float bb = gr_b1[n];
#pragma unroll
      for (int mt = 0; mt < 4; ++mt)
#pragma unroll
        for (int r = 0; r < 8; ++r) {
          const int row = mt * 16 + r + ((lane >> 4) << 3);
          h2s[(size_t)row * H_DIM + n] = geluf(acc[mt][t][r] + bb);
        }
    }
  }
  __syncthreads();

  // ---------------- logits = h2 @ gr_W2 + b2 ----------------
  for (int rr = 0; rr < 8; ++rr) {
    const int row = rowBase + rr;
    float d0 = 0.f;
#pragma unroll
    for (int i = 0; i < 12; ++i) {
      const int k = lane + 32 * i;
      d0 += h2s[(size_t)row * H_DIM + k] * gr_W2[k];
    }
    d0 = wsum(d0) + gr_b2[0];
    if (lane == 0) {
      logitsS[row] = d0;
      out[OFF_OK + (size_t)blk * ROWS + row] = d0;  // ori_key
    }
  }
  __syncthreads();

  // ------- softmax -> gaussian conv -> softmax -> interval search -------
  if (wv < 2) {
    const size_t gb = (size_t)blk * 2 + wv;
    const float lg = logitsS[wv * 32 + lane];
    // softmax over L (kp)
    float m = wmaxr(lg);
    float e = expf(lg - m);
    float p = e / wsum(e);
    // adaptive gaussian kernel from sigma
    const float sg = sigmaP[0];
    float kern[5], ks = 0.f;
#pragma unroll
    for (int i = 0; i < 5; ++i) {
      const float x = (float)(i - 2) / sg;
      kern[i] = expf(-0.5f * x * x);
      ks += kern[i];
    }
    const float inv = 1.f / ks;
    // zero-padded 5-tap conv via wave shuffles
    float sm = 0.f;
#pragma unroll
    for (int j = 0; j < 5; ++j) {
      const int src = lane + j - 2;
      const float pv = __shfl(p, src & 31, 32);
      sm += (src >= 0 && src < 32) ? kern[j] * inv * pv : 0.f;
    }
    // softmax over L (gs)
    float m2 = wmaxr(sm);
    float e2 = expf(sm - m2);
    const float gs = e2 / wsum(e2);
    out[OFF_GS + gb * L_DIM + lane] = gs;

    // broadcast all gs into every lane, then identical serial search
    float gsv[32];
#pragma unroll
    for (int i = 0; i < 32; ++i) gsv[i] = __shfl(gs, i, 32);
    int amax = 0;
    float bestv = gsv[0];
#pragma unroll
    for (int i = 1; i < 32; ++i)
      if (gsv[i] > bestv) { bestv = gsv[i]; amax = i; }
    float cs[33];
    cs[0] = 0.f;
#pragma unroll
    for (int i = 0; i < 32; ++i) cs[i + 1] = cs[i] + gsv[i];
    float bestS = -3.4e38f;
    int s0 = 0, w0 = 1;
    const int wins[3] = {1, 3, 5};
#pragma unroll
    for (int wi = 0; wi < 3; ++wi) {
      const int w = wins[wi];
      for (int st = 0; st <= 32 - w; ++st) {
        if (amax >= st && amax < st + w) {
          const float sc = cs[st + w] - cs[st];
          if (sc > bestS) { bestS = sc; s0 = st; w0 = w; }
        }
      }
    }
    const int e0 = s0 + w0;
    if (lane == 0) {
      out[OFF_MIDX + gb * 2 + 0] = (float)s0;
      out[OFF_MIDX + gb * 2 + 1] = (float)e0;
      out[OFF_ST + gb] = (float)s0 / 31.f;
      out[OFF_EN + gb] = (float)e0 / 31.f;
    }
    out[OFF_MASK + gb * L_DIM + lane] = (lane >= s0 && lane <= e0) ? 1.f : 0.f;
  }
}

// ---------------------------------------------------------------------------
extern "C" void kernel_launch(void* const* d_in, const int* in_sizes, int n_in,
                              void* d_out, int out_size, void* d_ws,
                              size_t ws_size, hipStream_t stream) {
  (void)in_sizes; (void)n_in; (void)out_size; (void)ws_size;
  const float* v       = (const float*)d_in[0];
  const float* qa      = (const float*)d_in[1];
  const float* vp_g    = (const float*)d_in[2];
  const float* vp_b    = (const float*)d_in[3];
  const float* vp_W    = (const float*)d_in[4];
  const float* vp_bias = (const float*)d_in[5];
  const float* qp_g    = (const float*)d_in[6];
  const float* qp_b    = (const float*)d_in[7];
  const float* qp_W    = (const float*)d_in[8];
  const float* qp_bias = (const float*)d_in[9];
  const float* gr_g    = (const float*)d_in[10];
  const float* gr_b    = (const float*)d_in[11];
  const float* gr_W1   = (const float*)d_in[12];
  const float* gr_b1   = (const float*)d_in[13];
  const float* gr_W2   = (const float*)d_in[14];
  const float* gr_b2   = (const float*)d_in[15];
  const float* sigma   = (const float*)d_in[16];
  float* out = (float*)d_out;

  // workspace layout (f16): vp_W sw | qp_W sw | gr_W1 sw | qa2
  char* ws = (char*)d_ws;
  _Float16* bswVP = (_Float16*)(ws);                  // 768*768*2 = 1179648 B
  _Float16* bswQP = (_Float16*)(ws + 1179648);        // 768*768*2
  _Float16* bswG1 = (_Float16*)(ws + 2359296);        // 768*384*2 =  589824 B
  _Float16* qa2ws = (_Float16*)(ws + 2949120);        // 4096*768*2

  // one-time weight convert+swizzle into WMMA B-operand layout
  swizzleWKernel<<<144, 256, 0, stream>>>(vp_W, bswVP, 768, 768);   // 1152 tiles
  swizzleWKernel<<<144, 256, 0, stream>>>(qp_W, bswQP, 768, 768);
  swizzleWKernel<<<72, 256, 0, stream>>>(gr_W1, bswG1, 768, 384);   // 576 tiles

  // qa2 = gelu(LN(qa) @ qp_W + qp_bias), f16
  qaProjKernel<<<64, 256, ROWS * D_DIM * 2, stream>>>(qa, qp_g, qp_b, bswQP,
                                                      qp_bias, qa2ws);

  // fused main pipeline: 2 batches per block
  const size_t dynLds = 2 * (size_t)ROWS * D_DIM * 2 + ROWS * sizeof(float);
  groundingMainKernel<<<BS / 2, 256, dynLds, stream>>>(
      v, vp_g, vp_b, bswVP, vp_bias, gr_g, gr_b, bswG1, gr_b1, gr_W2, gr_b2,
      qa2ws, sigma, out);
}
